// RelationalTransformerUpdate_50551765074577
// MI455X (gfx1250) — matrix operand order repaired and checked by reading, not compiled, not run on hardware
//
#include <hip/hip_runtime.h>
#include <hip/hip_bf16.h>
#include <math.h>
#include <stdint.h>

// ---------------------------------------------------------------------------
// Problem constants (match reference)
// ---------------------------------------------------------------------------
#define L_C   8
#define D_C   256
#define H_C   8
#define DK_C  32
#define FF_C  1024
#define R_C   37
#define N_C   1024
#define EPS_C 1e-6f
#define KC    64     // K-chunk staged per TDM transfer (elements)

typedef __bf16 bf16_t;
typedef __attribute__((ext_vector_type(16))) __bf16 bf16x16;
typedef __attribute__((ext_vector_type(8)))  __bf16 bf16x8;
typedef __attribute__((ext_vector_type(8)))  float  f32x8;
typedef __attribute__((ext_vector_type(4)))  unsigned int u32x4;
typedef __attribute__((ext_vector_type(8)))  int i32x8;
typedef __attribute__((ext_vector_type(4)))  int i32x4;

// ---------------------------------------------------------------------------
// WMMA helpers — layouts per CDNA5 ISA 7.12.2 (wave32)
// A 16x32 bf16: lane<16 -> row=lane, K {0..7,16..23}; lane>=16 -> row=lane-16, K {8..15,24..31}
// B 32x16 bf16: lane%16 -> col, lane<16 -> K {0..7,16..23}; lane>=16 -> K {8..15,24..31}
// C/D 16x16 f32: lane%16 -> col, vgpr r -> row r + 8*(lane/16)
// ---------------------------------------------------------------------------
__device__ __forceinline__ bf16x16 cat8(bf16x8 lo, bf16x8 hi) {
  bf16x16 r;
#pragma unroll
  for (int i = 0; i < 8; ++i) { r[i] = lo[i]; r[8 + i] = hi[i]; }
  return r;
}

__device__ __forceinline__ f32x8 wmma_bf16(bf16x16 a, bf16x16 b, f32x8 c) {
  return __builtin_amdgcn_wmma_f32_16x16x32_bf16(false, a, false, b, (short)0, c,
                                                 false, false);
}

// A row-major [*, lda]; tile origin (m0, k0). Works on global or LDS pointers.
__device__ __forceinline__ bf16x16 load_a_frag(const bf16_t* A, int lda, int m0,
                                               int k0, int lane) {
  int row = m0 + (lane & 15);
  int kb  = k0 + ((lane >> 4) ? 8 : 0);
  const bf16_t* p = A + (size_t)row * lda + kb;
  bf16x8 lo = *(const bf16x8*)p;
  bf16x8 hi = *(const bf16x8*)(p + 16);
  return cat8(lo, hi);
}

// Bt stores B transposed: Bt[n][k], row-major [*, ldb]; tile origin (n0, k0)
__device__ __forceinline__ bf16x16 load_b_frag(const bf16_t* Bt, int ldb, int n0,
                                               int k0, int lane) {
  int col = n0 + (lane & 15);
  int kb  = k0 + ((lane >> 4) ? 8 : 0);
  const bf16_t* p = Bt + (size_t)col * ldb + kb;
  bf16x8 lo = *(const bf16x8*)p;
  bf16x8 hi = *(const bf16x8*)(p + 16);
  return cat8(lo, hi);
}

// ---------------------------------------------------------------------------
// Tensor Data Mover: 2D tile (tile_rows x tile_k elements, 2-byte data) from
// global (row stride = row_stride_elems) into LDS at lds_off.
// D# layout per CDNA5 ISA.
// ---------------------------------------------------------------------------
__device__ __forceinline__ void tdm_issue(u32x4 g0, i32x8 g1) {
  i32x4 z4 = {0, 0, 0, 0};
#if defined(__clang_major__) && (__clang_major__ >= 23)
  i32x8 z8 = {0, 0, 0, 0, 0, 0, 0, 0};
  __builtin_amdgcn_tensor_load_to_lds(g0, g1, z4, z4, z8, 0);
#else
  __builtin_amdgcn_tensor_load_to_lds(g0, g1, z4, z4, 0);
#endif
}

__device__ __forceinline__ void tdm_load_2d(unsigned lds_off, const void* gptr,
                                            unsigned tile_k, unsigned tile_rows,
                                            unsigned row_stride_elems,
                                            unsigned tensor_rows) {
  unsigned long long ga = (unsigned long long)(uintptr_t)gptr;
  u32x4 g0;
  g0[0] = 1u;                                   // count=1 (user descriptor)
  g0[1] = lds_off;                              // LDS byte address
  g0[2] = (unsigned)(ga & 0xFFFFFFFFu);         // global_addr[31:0]
  g0[3] = (unsigned)((ga >> 32) & 0x1FFFFFFu) | (2u << 30);  // addr[56:32], type=2
  i32x8 g1;
  unsigned td0 = row_stride_elems;              // tensor dim0 (elements)
  unsigned td1 = tensor_rows;                   // tensor dim1 (rows)
  g1[0] = (int)(1u << 16);                      // data_size=2B; no pad/iter/multicast
  g1[1] = (int)((td0 & 0xFFFFu) << 16);         // tensor_dim0[15:0]
  g1[2] = (int)((td0 >> 16) | ((td1 & 0xFFFFu) << 16));  // dim0 hi | dim1 lo
  g1[3] = (int)((td1 >> 16) | ((tile_k & 0xFFFFu) << 16));  // dim1 hi | tile_dim0
  g1[4] = (int)(tile_rows & 0xFFFFu);           // tile_dim1 (tile_dim2=0)
  g1[5] = (int)row_stride_elems;                // tensor_dim0_stride[31:0]
  g1[6] = 0;                                    // stride0 hi | stride1 lo
  g1[7] = 0;
  tdm_issue(g0, g1);
}

// ---------------------------------------------------------------------------
// Weight convert + transpose: in f32 [Lb][K][Nc] -> out bf16 [Lb][Nc][K]
// ---------------------------------------------------------------------------
__global__ void tconv_kernel(const float* __restrict__ in, bf16_t* __restrict__ out,
                             int Lb, int K, int Nc) {
  size_t idx = (size_t)blockIdx.x * blockDim.x + threadIdx.x;
  size_t per = (size_t)K * Nc;
  size_t tot = (size_t)Lb * per;
  if (idx >= tot) return;
  size_t l = idx / per, rem = idx % per;
  size_t kk = rem / Nc, nn = rem % Nc;
  out[l * per + nn * K + kk] = (bf16_t)in[idx];
}

// bf16 transpose: in [M][Nc] -> out [Nc][M]   (V -> Vt for PV matmul B-frags)
__global__ void tbf_kernel(const bf16_t* __restrict__ in, bf16_t* __restrict__ out,
                           int M, int Nc) {
  int idx = blockIdx.x * blockDim.x + threadIdx.x;
  if (idx >= M * Nc) return;
  int m = idx / Nc, n = idx % Nc;
  out[(size_t)n * M + m] = in[idx];
}

// ---------------------------------------------------------------------------
// LayerNorm (unbiased std, ddof=1): one row per 256-thread block
// ---------------------------------------------------------------------------
__global__ __launch_bounds__(256) void ln_kernel(const float* __restrict__ x,
                                                 const float* __restrict__ g,
                                                 const float* __restrict__ b,
                                                 bf16_t* outB, float* outF) {
  __shared__ float red[256];
  int row = blockIdx.x, t = threadIdx.x;
  float v = x[(size_t)row * D_C + t];
  red[t] = v;
  __syncthreads();
#pragma unroll
  for (int s = 128; s > 0; s >>= 1) {
    if (t < s) red[t] += red[t + s];
    __syncthreads();
  }
  float mu = red[0] * (1.0f / D_C);
  __syncthreads();
  float d = v - mu;
  red[t] = d * d;
  __syncthreads();
#pragma unroll
  for (int s = 128; s > 0; s >>= 1) {
    if (t < s) red[t] += red[t + s];
    __syncthreads();
  }
  float sd = sqrtf(red[0] * (1.0f / (D_C - 1)));
  float y = g[t] * d / (sd + EPS_C) + b[t];
  if (outB) outB[(size_t)row * D_C + t] = (bf16_t)y;
  else      outF[(size_t)row * D_C + t] = y;
}

// ---------------------------------------------------------------------------
// qrel[h,i,r] = sum_dk q[i, h*DK+dk] * rel_k[r, dk]
// ---------------------------------------------------------------------------
__global__ void qrel_kernel(const bf16_t* __restrict__ q,
                            const float* __restrict__ relk,
                            float* __restrict__ qrel) {
  int idx = blockIdx.x * blockDim.x + threadIdx.x;
  if (idx >= H_C * N_C * R_C) return;
  int r = idx % R_C;
  int rem = idx / R_C;
  int i = rem % N_C;
  int h = rem / N_C;
  const bf16_t* qp = q + (size_t)i * D_C + h * DK_C;
  const float* rp = relk + r * DK_C;
  float s = 0.f;
#pragma unroll
  for (int d = 0; d < DK_C; ++d) s += (float)qp[d] * rp[d];
  qrel[idx] = s;
}

// ---------------------------------------------------------------------------
// WMMA GEMM with TDM double-buffered LDS staging.
//   C[M,Nc] = A[M,K](bf16) * Bt[Nc,K]^T(bf16) + bias
// mode 0: store bf16   mode 1: f32 accumulate (residual +=)   mode 2: ReLU->bf16
// WG = 256 thr (8 waves, 4M x 2N); wave tile 32x32 (4 WMMAs / 32-K step).
// WG tile 128x64; per chunk TDM stages A 128xKC and B 64xKC into LDS.
// ---------------------------------------------------------------------------
__global__ __launch_bounds__(256) void gemm_kernel(
    const bf16_t* __restrict__ A, const bf16_t* __restrict__ Bt,
    const float* __restrict__ bias, bf16_t* __restrict__ outB,
    float* __restrict__ outAcc, int M, int Nc, int K, int mode) {
  __shared__ bf16_t As[2][128 * KC];
  __shared__ bf16_t Bs[2][64 * KC];

  int tid = threadIdx.x;
  int lane = tid & 31;
  int w  = tid >> 5;
  int wm = w & 3, wn = w >> 2;
  int m0g = blockIdx.y * 128;
  int n0g = blockIdx.x * 64;
  int m0l = wm * 32, n0l = wn * 32;

  const int nch = K / KC;
  const bool issuer = (tid < 32);  // wave 0 drives the TDM

  if (issuer) {
    tdm_load_2d((unsigned)(uintptr_t)&As[0][0], A + (size_t)m0g * K, KC, 128, K, M);
    tdm_load_2d((unsigned)(uintptr_t)&Bs[0][0], Bt + (size_t)n0g * K, KC, 64, K, Nc);
  }

  f32x8 acc00 = {}, acc01 = {}, acc10 = {}, acc11 = {};
  for (int c = 0; c < nch; ++c) {
    if (issuer && (c + 1 < nch)) {
      int nb = (c + 1) & 1;
      tdm_load_2d((unsigned)(uintptr_t)&As[nb][0],
                  A + (size_t)m0g * K + (size_t)(c + 1) * KC, KC, 128, K, M);
      tdm_load_2d((unsigned)(uintptr_t)&Bs[nb][0],
                  Bt + (size_t)n0g * K + (size_t)(c + 1) * KC, KC, 64, K, Nc);
    }
    if (issuer) {
      if (c + 1 < nch) __builtin_amdgcn_s_wait_tensorcnt(2);  // chunk c landed
      else             __builtin_amdgcn_s_wait_tensorcnt(0);
    }
    __syncthreads();

    const bf16_t* Ab = As[c & 1];
    const bf16_t* Bb = Bs[c & 1];
#pragma unroll
    for (int k0 = 0; k0 < KC; k0 += 32) {
      bf16x16 a0 = load_a_frag(Ab, KC, m0l, k0, lane);
      bf16x16 a1 = load_a_frag(Ab, KC, m0l + 16, k0, lane);
      bf16x16 b0 = load_b_frag(Bb, KC, n0l, k0, lane);
      bf16x16 b1 = load_b_frag(Bb, KC, n0l + 16, k0, lane);
      acc00 = wmma_bf16(a0, b0, acc00);
      acc01 = wmma_bf16(a0, b1, acc01);
      acc10 = wmma_bf16(a1, b0, acc10);
      acc11 = wmma_bf16(a1, b1, acc11);
    }
    __syncthreads();  // buffer may be overwritten by next iteration's TDM
  }

  int hi = lane >> 4, ln = lane & 15;
  int na = n0g + n0l + ln;
  int nb = na + 16;
  float ba = bias[na], bb = bias[nb];
#pragma unroll
  for (int r = 0; r < 8; ++r) {
    int ma = m0g + m0l + r + 8 * hi;
    int mb = ma + 16;
    float v00 = acc00[r] + ba, v01 = acc01[r] + bb;
    float v10 = acc10[r] + ba, v11 = acc11[r] + bb;
    if (mode == 0) {
      outB[(size_t)ma * Nc + na] = (bf16_t)v00;
      outB[(size_t)ma * Nc + nb] = (bf16_t)v01;
      outB[(size_t)mb * Nc + na] = (bf16_t)v10;
      outB[(size_t)mb * Nc + nb] = (bf16_t)v11;
    } else if (mode == 1) {
      outAcc[(size_t)ma * Nc + na] += v00;
      outAcc[(size_t)ma * Nc + nb] += v01;
      outAcc[(size_t)mb * Nc + na] += v10;
      outAcc[(size_t)mb * Nc + nb] += v11;
    } else {
      outB[(size_t)ma * Nc + na] = (bf16_t)fmaxf(v00, 0.f);
      outB[(size_t)ma * Nc + nb] = (bf16_t)fmaxf(v01, 0.f);
      outB[(size_t)mb * Nc + na] = (bf16_t)fmaxf(v10, 0.f);
      outB[(size_t)mb * Nc + nb] = (bf16_t)fmaxf(v11, 0.f);
    }
  }
}

// ---------------------------------------------------------------------------
// Relation-aware flash attention.
// One wave per (head, 16-row tile); streams 32-key blocks.
//  scores = (Q K^T + qrel[i, rel(i,j)]) * scale   -> online softmax
//  out    = (P V + prel @ rel_v) / rowsum         (prel binned in LDS)
// ---------------------------------------------------------------------------
__global__ __launch_bounds__(32) void attn_kernel(
    const bf16_t* __restrict__ q, const bf16_t* __restrict__ k,
    const bf16_t* __restrict__ vt, const float* __restrict__ qrel,
    const int* __restrict__ relations, const float* __restrict__ relv,
    bf16_t* __restrict__ o, float scale) {
  const int lane = threadIdx.x;
  const int h  = blockIdx.y;
  const int i0 = blockIdx.x * 16;
  const int hi = lane >> 4;
  const int ln = lane & 15;

  __shared__ float p_lds[16][32];
  __shared__ float prel[16][R_C];
  __shared__ float alpha_lds[16];

  for (int idx = lane; idx < 16 * R_C; idx += 32) ((float*)prel)[idx] = 0.f;
  __syncthreads();

  bf16x16 qfrag = load_a_frag(q, D_C, i0, h * DK_C, lane);
  f32x8 acc0 = {}, acc1 = {};
  float m_r[8], s_r[8];
#pragma unroll
  for (int r = 0; r < 8; ++r) { m_r[r] = -1e30f; s_r[r] = 0.f; }

  for (int j0 = 0; j0 < N_C; j0 += 32) {
    if (j0 + 32 < N_C) {  // hint next K/V blocks toward the caches
      __builtin_prefetch(k + (size_t)(j0 + 32 + ln) * D_C + h * DK_C, 0, 1);
      __builtin_prefetch(vt + (size_t)(h * DK_C + ln) * N_C + j0 + 32, 0, 1);
    }
    // ---- scores: Q (16x32) x K^T block (32 x 32) -> two 16x16 f32 frags
    bf16x16 bk0 = load_b_frag(k, D_C, j0,      h * DK_C, lane);
    bf16x16 bk1 = load_b_frag(k, D_C, j0 + 16, h * DK_C, lane);
    f32x8 z = {};
    f32x8 s0 = wmma_bf16(qfrag, bk0, z);
    f32x8 s1 = wmma_bf16(qfrag, bk1, z);

    float sc0[8], sc1[8];
    int rl0[8], rl1[8];
#pragma unroll
    for (int r = 0; r < 8; ++r) {
      int row = i0 + r + 8 * hi;
      int j   = j0 + ln;
      rl0[r] = relations[(size_t)row * N_C + j];
      rl1[r] = relations[(size_t)row * N_C + j + 16];
      const float* qr = qrel + ((size_t)h * N_C + row) * R_C;
      sc0[r] = (s0[r] + qr[rl0[r]]) * scale;
      sc1[r] = (s1[r] + qr[rl1[r]]) * scale;
    }

    // ---- online softmax update (cross-lane reduce within 16-lane groups)
    float alpha[8];
#pragma unroll
    for (int r = 0; r < 8; ++r) {
      float vm = fmaxf(sc0[r], sc1[r]);
      vm = fmaxf(vm, __shfl_xor(vm, 1, 16));
      vm = fmaxf(vm, __shfl_xor(vm, 2, 16));
      vm = fmaxf(vm, __shfl_xor(vm, 4, 16));
      vm = fmaxf(vm, __shfl_xor(vm, 8, 16));
      float mnew = fmaxf(m_r[r], vm);
      alpha[r] = __expf(m_r[r] - mnew);
      sc0[r] = __expf(sc0[r] - mnew);
      sc1[r] = __expf(sc1[r] - mnew);
      float rs = sc0[r] + sc1[r];
      rs += __shfl_xor(rs, 1, 16);
      rs += __shfl_xor(rs, 2, 16);
      rs += __shfl_xor(rs, 4, 16);
      rs += __shfl_xor(rs, 8, 16);
      s_r[r] = s_r[r] * alpha[r] + rs;
      m_r[r] = mnew;
      acc0[r] *= alpha[r];
      acc1[r] *= alpha[r];
    }
    if (lane == 0) {
#pragma unroll
      for (int r = 0; r < 8; ++r) alpha_lds[r] = alpha[r];
    }
    if (lane == 16) {
#pragma unroll
      for (int r = 0; r < 8; ++r) alpha_lds[8 + r] = alpha[r];
    }
    __syncthreads();

    // ---- rescale relation bins, then scatter-add this block's p values
    for (int idx = lane; idx < 16 * R_C; idx += 32)
      ((float*)prel)[idx] *= alpha_lds[idx / R_C];
    __syncthreads();
#pragma unroll
    for (int r = 0; r < 8; ++r) {
      int row_l = r + 8 * hi;
      atomicAdd(&prel[row_l][rl0[r]], sc0[r]);
      atomicAdd(&prel[row_l][rl1[r]], sc1[r]);
      p_lds[row_l][ln]      = sc0[r];
      p_lds[row_l][ln + 16] = sc1[r];
    }
    __syncthreads();

    // ---- P tile (C-layout -> A-layout via LDS), then PV WMMAs
    bf16x16 pfrag;
    {
      int row = ln;
      int kb  = hi ? 8 : 0;
#pragma unroll
      for (int t = 0; t < 8; ++t) {
        pfrag[t]     = (bf16_t)p_lds[row][kb + t];
        pfrag[8 + t] = (bf16_t)p_lds[row][kb + 16 + t];
      }
    }
    bf16x16 bv0 = load_b_frag(vt, N_C, h * DK_C,      j0, lane);
    bf16x16 bv1 = load_b_frag(vt, N_C, h * DK_C + 16, j0, lane);
    acc0 = wmma_bf16(pfrag, bv0, acc0);
    acc1 = wmma_bf16(pfrag, bv1, acc1);
    __syncthreads();  // before next iteration overwrites p_lds
  }

  // ---- epilogue: add prel @ rel_v, normalize, store bf16
#pragma unroll
  for (int r = 0; r < 8; ++r) {
    int row_l = r + 8 * hi;
    float inv = 1.0f / s_r[r];
    float e0 = 0.f, e1 = 0.f;
    for (int rr = 0; rr < R_C; ++rr) {
      float pr = prel[row_l][rr];
      e0 += pr * relv[rr * DK_C + ln];
      e1 += pr * relv[rr * DK_C + 16 + ln];
    }
    size_t base = (size_t)(i0 + row_l) * D_C + h * DK_C;
    o[base + ln]      = (bf16_t)((acc0[r] + e0) * inv);
    o[base + 16 + ln] = (bf16_t)((acc1[r] + e1) * inv);
  }
}

// ---------------------------------------------------------------------------
// Host-side orchestration
// ---------------------------------------------------------------------------
extern "C" void kernel_launch(void* const* d_in, const int* in_sizes, int n_in,
                              void* d_out, int out_size, void* d_ws, size_t ws_size,
                              hipStream_t stream) {
  (void)in_sizes; (void)n_in; (void)out_size; (void)ws_size;
  const float* x_in   = (const float*)d_in[0];
  const int*   rels   = (const int*)d_in[1];
  const float* Wq = (const float*)d_in[2];  const float* bq = (const float*)d_in[3];
  const float* Wk = (const float*)d_in[4];  const float* bk = (const float*)d_in[5];
  const float* Wv = (const float*)d_in[6];  const float* bv = (const float*)d_in[7];
  const float* Wo = (const float*)d_in[8];  const float* bo = (const float*)d_in[9];
  const float* rel_k = (const float*)d_in[10];
  const float* rel_v = (const float*)d_in[11];
  const float* W1 = (const float*)d_in[12]; const float* b1 = (const float*)d_in[13];
  const float* W2 = (const float*)d_in[14]; const float* b2 = (const float*)d_in[15];
  const float* ln1_g = (const float*)d_in[16]; const float* ln1_b = (const float*)d_in[17];
  const float* ln2_g = (const float*)d_in[18]; const float* ln2_b = (const float*)d_in[19];
  const float* lnf_g = (const float*)d_in[20]; const float* lnf_b = (const float*)d_in[21];

  char* ws = (char*)d_ws;
  size_t off = 0;
  auto alloc = [&](size_t bytes) -> void* {
    void* p = ws + off;
    off += (bytes + 255) & ~(size_t)255;
    return p;
  };
  bf16_t* Wqt = (bf16_t*)alloc((size_t)L_C * D_C * D_C * 2);
  bf16_t* Wkt = (bf16_t*)alloc((size_t)L_C * D_C * D_C * 2);
  bf16_t* Wvt = (bf16_t*)alloc((size_t)L_C * D_C * D_C * 2);
  bf16_t* Wot = (bf16_t*)alloc((size_t)L_C * D_C * D_C * 2);
  bf16_t* W1t = (bf16_t*)alloc((size_t)L_C * D_C * FF_C * 2);
  bf16_t* W2t = (bf16_t*)alloc((size_t)L_C * FF_C * D_C * 2);
  float*  x_f   = (float*)alloc((size_t)N_C * D_C * 4);
  bf16_t* xn_bf = (bf16_t*)alloc((size_t)N_C * D_C * 2);
  bf16_t* q_bf  = (bf16_t*)alloc((size_t)N_C * D_C * 2);
  bf16_t* k_bf  = (bf16_t*)alloc((size_t)N_C * D_C * 2);
  bf16_t* v_bf  = (bf16_t*)alloc((size_t)N_C * D_C * 2);
  bf16_t* vt_bf = (bf16_t*)alloc((size_t)N_C * D_C * 2);
  bf16_t* o_bf  = (bf16_t*)alloc((size_t)N_C * D_C * 2);
  bf16_t* h_bf  = (bf16_t*)alloc((size_t)N_C * FF_C * 2);
  float*  qrel_f = (float*)alloc((size_t)H_C * N_C * R_C * 4);

  const float scale = 0.17677669529663687f;  // 1/sqrt(DK)

  // ---- one-time weight convert + transpose (bf16, column-major)
  {
    size_t tot = (size_t)L_C * D_C * D_C;
    int blk = (int)((tot + 255) / 256);
    tconv_kernel<<<blk, 256, 0, stream>>>(Wq, Wqt, L_C, D_C, D_C);
    tconv_kernel<<<blk, 256, 0, stream>>>(Wk, Wkt, L_C, D_C, D_C);
    tconv_kernel<<<blk, 256, 0, stream>>>(Wv, Wvt, L_C, D_C, D_C);
    tconv_kernel<<<blk, 256, 0, stream>>>(Wo, Wot, L_C, D_C, D_C);
    size_t tf = (size_t)L_C * D_C * FF_C;
    int blkf = (int)((tf + 255) / 256);
    tconv_kernel<<<blkf, 256, 0, stream>>>(W1, W1t, L_C, D_C, FF_C);
    tconv_kernel<<<blkf, 256, 0, stream>>>(W2, W2t, L_C, FF_C, D_C);
  }

  // ---- residual stream starts as the input
  hipMemcpyAsync(x_f, x_in, (size_t)N_C * D_C * 4, hipMemcpyDeviceToDevice, stream);

  dim3 gD(D_C / 64, N_C / 128);
  dim3 gF(FF_C / 64, N_C / 128);

  for (int l = 0; l < L_C; ++l) {
    // --- attention sublayer
    ln_kernel<<<N_C, 256, 0, stream>>>(x_f, ln1_g + l * D_C, ln1_b + l * D_C,
                                       xn_bf, nullptr);
    gemm_kernel<<<gD, 256, 0, stream>>>(xn_bf, Wqt + (size_t)l * D_C * D_C,
                                        bq + l * D_C, q_bf, nullptr, N_C, D_C, D_C, 0);
    gemm_kernel<<<gD, 256, 0, stream>>>(xn_bf, Wkt + (size_t)l * D_C * D_C,
                                        bk + l * D_C, k_bf, nullptr, N_C, D_C, D_C, 0);
    gemm_kernel<<<gD, 256, 0, stream>>>(xn_bf, Wvt + (size_t)l * D_C * D_C,
                                        bv + l * D_C, v_bf, nullptr, N_C, D_C, D_C, 0);
    tbf_kernel<<<(N_C * D_C + 255) / 256, 256, 0, stream>>>(v_bf, vt_bf, N_C, D_C);
    qrel_kernel<<<(H_C * N_C * R_C + 255) / 256, 256, 0, stream>>>(
        q_bf, rel_k + (size_t)l * R_C * DK_C, qrel_f);
    attn_kernel<<<dim3(N_C / 16, H_C), 32, 0, stream>>>(
        q_bf, k_bf, vt_bf, qrel_f, rels, rel_v + (size_t)l * R_C * DK_C, o_bf, scale);
    gemm_kernel<<<gD, 256, 0, stream>>>(o_bf, Wot + (size_t)l * D_C * D_C,
                                        bo + l * D_C, nullptr, x_f, N_C, D_C, D_C, 1);

    // --- FFN sublayer
    ln_kernel<<<N_C, 256, 0, stream>>>(x_f, ln2_g + l * D_C, ln2_b + l * D_C,
                                       xn_bf, nullptr);
    gemm_kernel<<<gF, 256, 0, stream>>>(xn_bf, W1t + (size_t)l * D_C * FF_C,
                                        b1 + l * FF_C, h_bf, nullptr, N_C, FF_C, D_C, 2);
    gemm_kernel<<<gD, 256, 0, stream>>>(h_bf, W2t + (size_t)l * FF_C * D_C,
                                        b2 + l * D_C, nullptr, x_f, N_C, D_C, FF_C, 1);
  }

  // ---- final layernorm -> f32 output
  ln_kernel<<<N_C, 256, 0, stream>>>(x_f, lnf_g, lnf_b, nullptr, (float*)d_out);
}
